// SparseConvTranspose2d_65309272703018
// MI455X (gfx1250) — matrix operand order repaired and checked by reading, not compiled, not run on hardware
//
#include <hip/hip_runtime.h>

typedef __attribute__((ext_vector_type(8)))  __bf16 v8bf;
typedef __attribute__((ext_vector_type(16))) __bf16 v16bf;
typedef __attribute__((ext_vector_type(8)))  float  v8f;

#define X_COLS   66    // cols staged per row: [Xbase-1, Xbase+64]
#define X_STRIDE 72    // bf16 elems per col entry (pad 64 -> 72: 144B, 16B-aligned)
#define B_STRIDE 264   // 32 cout rows of 256 K, padded (528B rows)
#define C_STRIDE 65    // f32 epilogue transpose rows

__device__ __forceinline__ unsigned short f2bf(float f) {
    unsigned int u = __float_as_uint(f);
    unsigned int r = u + 0x7FFFu + ((u >> 16) & 1u);   // round-to-nearest-even
    return (unsigned short)(r >> 16);
}

// ---- prep: pack per-parity, K-ordered bf16 B matrix into d_ws -------------
// Bw[par][n][k], k = t*64+ci, t = 2i+j, kh = 2i+1-ph, kw = 2j+1-pw
__global__ __launch_bounds__(256)
void pack_b_kernel(const float* __restrict__ w, unsigned short* __restrict__ Bw) {
    const int idx = blockIdx.x * 256 + threadIdx.x;       // < 4*32*256
    const int par = idx >> 13;
    const int n   = (idx >> 8) & 31;
    const int k   = idx & 255;
    const int ph = par >> 1, pw = par & 1;
    const int t = k >> 6, ci = k & 63;
    const int i = t >> 1, j = t & 1;
    const int kh = 2 * i + 1 - ph;
    const int kw = 2 * j + 1 - pw;
    Bw[idx] = f2bf(w[((ci * 32 + n) * 4 + kh) * 4 + kw]);
}

__global__ __launch_bounds__(128)
void sct2d_wmma_kernel(const float* __restrict__ x, const float* __restrict__ w,
                       const float* __restrict__ bias, float* __restrict__ out,
                       const unsigned short* __restrict__ Bw, int useBw) {
    // raw input rows, bf16, layout [i(2)][col(66)][ci(64)] padded to 72
    __shared__ __align__(16) unsigned short sX[2 * X_COLS * X_STRIDE]; // 19008 B
    __shared__ __align__(16) unsigned short sB[32 * B_STRIDE];         // 16896 B

    const int tid = threadIdx.x;
    const int xt  = blockIdx.x & 1;
    const int Y   = blockIdx.x >> 1;       // 0..127
    const int ph  = blockIdx.y >> 1;
    const int pw  = blockIdx.y & 1;
    const int b   = blockIdx.z;
    const int Xbase = xt << 6;

    // ---- stage raw x rows: sX[i][c][ci] = x[b, ci, Y+ph-i, Xbase+c-1]
    for (int e = tid; e < 2 * X_COLS * 64; e += 128) {     // 8448 elems, 66 iters
        const int i  = e / (X_COLS * 64);
        const int r  = e - i * (X_COLS * 64);
        const int ci = r / X_COLS;
        const int c  = r - ci * X_COLS;
        const int iy = Y + ph - i;
        const int ix = Xbase + c - 1;
        float v = 0.0f;
        if ((unsigned)iy < 128u && (unsigned)ix < 128u)
            v = x[(((b << 6) + ci) * 128 + iy) * 128 + ix];
        sX[(i * X_COLS + c) * X_STRIDE + ci] = f2bf(v);
    }
    // ---- stage B tile (column-major rows of K)
    const int par = (ph << 1) | pw;
    if (useBw) {                                           // coalesced 16B copies
        const uint4* src = (const uint4*)(Bw + par * 8192);
        for (int e = tid; e < 1024; e += 128) {            // 8192 ushorts
            const int n = e >> 5, koff = (e & 31) << 3;    // 8-elem chunks
            *(uint4*)(sB + n * B_STRIDE + koff) = src[(n << 5) + (e & 31)];
        }
    } else {                                               // fallback: direct from w
        for (int e = tid; e < 32 * 256; e += 128) {
            const int n = e >> 8, k = e & 255;
            const int t = k >> 6, ci = k & 63;
            const int i = t >> 1, j = t & 1;
            const int kh = 2 * i + 1 - ph;
            const int kw = 2 * j + 1 - pw;
            sB[n * B_STRIDE + k] = f2bf(w[((ci * 32 + n) * 4 + kh) * 4 + kw]);
        }
    }
    __syncthreads();

    // ---- GEMM: per wave 16(px) x 32(cout) x 256(K) -> 16 WMMA
    const int wid  = tid >> 5;
    const int lane = tid & 31;
    const int g    = lane >> 4;
    const int l16  = lane & 15;
    const int p    = wid * 16 + l16;                       // wave-local pixel
    // col for tap j: c_j = p + dx_j + 1, dx_j = pw - j  ->  c_j = p + pw + 1 - j
    const int c0 = p + pw + 1;                             // j = 0
    const int c1 = p + pw;                                 // j = 1
    const unsigned short* aT[4] = {                        // t = 2i + j
        sX + (0 * X_COLS + c0) * X_STRIDE + g * 8,         // t0: i=0 j=0
        sX + (0 * X_COLS + c1) * X_STRIDE + g * 8,         // t1: i=0 j=1
        sX + (1 * X_COLS + c0) * X_STRIDE + g * 8,         // t2: i=1 j=0
        sX + (1 * X_COLS + c1) * X_STRIDE + g * 8,         // t3: i=1 j=1
    };
    const unsigned short* b0B = sB + l16 * B_STRIDE + g * 8;
    const unsigned short* b1B = sB + (16 + l16) * B_STRIDE + g * 8;

    v8f acc0 = {}; v8f acc1 = {};
#pragma unroll
    for (int s = 0; s < 8; ++s) {
        const int k0  = s << 5;
        const int t   = s >> 1;                            // compile-time per iter
        const int cio = (s & 1) << 5;                      // ci offset (g*8 in base)
        v8bf alo = *(const v8bf*)(aT[t] + cio);
        v8bf ahi = *(const v8bf*)(aT[t] + cio + 16);
        v16bf a  = __builtin_shufflevector(alo, ahi, 0,1,2,3,4,5,6,7,8,9,10,11,12,13,14,15);
        v8bf b0l = *(const v8bf*)(b0B + k0);
        v8bf b0h = *(const v8bf*)(b0B + k0 + 16);
        v16bf bb0 = __builtin_shufflevector(b0l, b0h, 0,1,2,3,4,5,6,7,8,9,10,11,12,13,14,15);
        v8bf b1l = *(const v8bf*)(b1B + k0);
        v8bf b1h = *(const v8bf*)(b1B + k0 + 16);
        v16bf bb1 = __builtin_shufflevector(b1l, b1h, 0,1,2,3,4,5,6,7,8,9,10,11,12,13,14,15);
        acc0 = __builtin_amdgcn_wmma_f32_16x16x32_bf16(false, a, false, bb0,
                                                       (short)0, acc0, false, false);
        acc1 = __builtin_amdgcn_wmma_f32_16x16x32_bf16(false, a, false, bb1,
                                                       (short)0, acc1, false, false);
    }

    // ---- epilogue: LDS transpose (reuse sX), bias, store
    __syncthreads();
    float* Ct = (float*)sX;                                // 32*65*4 = 8320 B
#pragma unroll
    for (int r = 0; r < 8; ++r) {                          // D: lane(g,n), VGPR r -> M=8g+r
        const int px = wid * 16 + (g << 3) + r;
        Ct[l16 * C_STRIDE + px]        = acc0[r];
        Ct[(16 + l16) * C_STRIDE + px] = acc1[r];
    }
    __syncthreads();

    const int oh = (Y << 1) + ph;
    for (int e = tid; e < 64 * 32; e += 128) {
        const int px = e & 63;
        const int co = e >> 6;
        const float v = Ct[co * C_STRIDE + px] + bias[co];
        const int ow  = ((Xbase + px) << 1) + pw;
        out[(((b * 32 + co) * 256) + oh) * 256 + ow] = v;
    }
}

extern "C" void kernel_launch(void* const* d_in, const int* in_sizes, int n_in,
                              void* d_out, int out_size, void* d_ws, size_t ws_size,
                              hipStream_t stream) {
    const float* x    = (const float*)d_in[0];   // (16,64,128,128)
    const float* wgt  = (const float*)d_in[1];   // (64,32,4,4)
    const float* bias = (const float*)d_in[2];   // (32,)
    float* out = (float*)d_out;                  // (16,32,256,256)

    const int useBw = (ws_size >= (size_t)(4 * 32 * 256 * 2)) ? 1 : 0;
    unsigned short* Bw = (unsigned short*)d_ws;

    if (useBw) {
        pack_b_kernel<<<dim3(128), dim3(256), 0, stream>>>(wgt, Bw);
    }
    dim3 grid(256 /* Y(128) x Xtile(2) */, 4 /* parities */, 16 /* batch */);
    sct2d_wmma_kernel<<<grid, dim3(128), 0, stream>>>(x, wgt, bias, out, Bw, useBw);
}
